// PointGNNCon_83863531422139
// MI455X (gfx1250) — compile-verified, matrix operand order checked by reading, not run on hardware
//
#include <hip/hip_runtime.h>
#include <math.h>

// ---------------------------------------------------------------------------
// PointGNN forward for MI455X (gfx1250, wave32, WMMA 16x16x32 f16->f32).
//  * All GEMMs use v_wmma_f32_16x16x32_f16 (f16 operands, f32 accum).
//  * BatchNorm stats: f32 col-sum/sumsq via LDS-atomic block reduction then
//    global f32 atomics.
//  * segment_max: native global_atomic_max_num_f32 (inline asm, device scope).
//  * Edge-MLP BN handled with a two-pass (stats, then apply) recompute scheme
//    to avoid a 205MB H1 intermediate; node features fit in the 192MB L2 so
//    the random x[src] gathers are L2 hits.
//  * NOUT and leading dims are compile-time so GEMM epilogues are branch-free
//    with shift-based addressing (single addr calc + clause of stores).
// ---------------------------------------------------------------------------

typedef _Float16 f16;
typedef __attribute__((ext_vector_type(8)))  _Float16 v8h;
typedef __attribute__((ext_vector_type(16))) _Float16 v16h;
typedef __attribute__((ext_vector_type(8)))  float    v8f;

#define DEVINL __device__ __forceinline__

DEVINL v8f wmma_f16(v16h a, v16h b, v8f c) {
  // (neg_a, A, neg_b, B, c_mod, C, reuse_a, reuse_b)
  return __builtin_amdgcn_wmma_f32_16x16x32_f16(false, a, false, b, (short)0, c,
                                                false, false);
}

// A-fragment load, row-major source (global or LDS), 16x32 f16 tile.
// lanes 0-15: row=lane,     K = [kb..kb+7] and [kb+16..kb+23]
// lanes16-31: row=lane-16,  K = [kb+8..kb+15] and [kb+24..kb+31]
DEVINL v16h load_a(const f16* base, int ld, int kbase, int lane) {
  int row = lane & 15;
  int ko  = (lane >> 4) * 8;
  const f16* q = base + row * ld + kbase + ko;
  v8h lo = *(const v8h*)q;
  v8h hi = *(const v8h*)(q + 16);
  v16h a;
#pragma unroll
  for (int t = 0; t < 8; ++t) { a[t] = lo[t]; a[t + 8] = hi[t]; }
  return a;
}

// Device-scope float atomic max (matches the scope:SCOPE_DEV the compiler
// emits for its own global atomics; WGP default scope would not be coherent
// across the WGPs scattering into agg).
DEVINL void atomic_fmax_f32(float* addr, float v) {
  asm volatile("global_atomic_max_num_f32 %0, %1, off scope:SCOPE_DEV" ::"v"(
                   addr),
               "v"(v)
               : "memory");
}

// ---------------------------------------------------------------------------
// Weight packing: row-major f32 W[K][Nout] -> f16 WMMA-B fragments.
// Layout per (ntile, kchunk): 32 lanes x 16 contiguous f16 (mirrors A layout
// with (K,N) roles). perm!=0 reorders rows for the edge MLP f1 so A can be
// staged as [x(perm) | rel(3) | pad] while the reference order is [rel | x].
// ---------------------------------------------------------------------------
__global__ void k_pack(const float* __restrict__ W, int K, int Nout, int kc,
                       int nt, int perm, f16* __restrict__ dst) {
  int idx = blockIdx.x * blockDim.x + threadIdx.x;
  int total = kc * nt * 512;
  if (idx >= total) return;
  int i    = idx & 15;
  int lane = (idx >> 4) & 31;
  int rem  = idx >> 9;
  int kcc  = rem % kc;
  int ntt  = rem / kc;
  int col  = ntt * 16 + (lane & 15);
  int kb   = (lane >> 4) * 8;
  int k    = kcc * 32 + ((i < 8) ? (kb + i) : (kb + 16 + (i - 8)));
  int kk   = perm ? ((k < perm) ? (k + 3) : (k - perm)) : k;
  f16 v    = (f16)0.f;
  if (k < K && col < Nout) v = (f16)W[(size_t)kk * Nout + col];
  dst[idx] = v;
}

// ---------------------------------------------------------------------------
// Node GEMM: Y[nrows x NOUT] = Xf16[nrows x KC*32] * W + b, optional stats.
// One wave per 16-row tile, NT 16-col tiles, KC k-chunks of 32; NOUT is a
// compile-time constant so predicates fold and addressing is shift-based.
// ---------------------------------------------------------------------------
template <int KC, int NT, int NOUT>
__global__ __launch_bounds__(256) void k_gemm(
    const f16* __restrict__ X, const f16* __restrict__ Wp,
    const float* __restrict__ bias, float* __restrict__ Y, int nrows,
    float* __restrict__ statS, float* __restrict__ statQ, int doStats) {
  constexpr int LDX = KC * 32;
  int wave = blockIdx.x * (blockDim.x >> 5) + (threadIdx.x >> 5);
  int row0 = wave * 16;
  if (row0 >= nrows) return;
  int lane = threadIdx.x & 31;

  v16h A[KC];
#pragma unroll
  for (int k = 0; k < KC; ++k)
    A[k] = load_a(X + (size_t)row0 * LDX, LDX, k * 32, lane);

#pragma unroll
  for (int nt = 0; nt < NT; ++nt) {
    v8f acc;
#pragma unroll
    for (int v = 0; v < 8; ++v) acc[v] = 0.f;
#pragma unroll
    for (int k = 0; k < KC; ++k) {
      v16h B = *(const v16h*)(Wp + ((size_t)(nt * KC + k) * 32 + lane) * 16);
      acc = wmma_f16(A[k], B, acc);
    }
    int c   = nt * 16 + (lane & 15);
    bool ok = (NOUT % 16 == 0) ? true : (c < NOUT);
    float bb = ok ? bias[c] : 0.f;
    float yv[8];
    float s = 0.f, q = 0.f;
#pragma unroll
    for (int v = 0; v < 8; ++v) {
      yv[v] = acc[v] + bb;
      s += yv[v];
      q += yv[v] * yv[v];
    }
    if (ok) {
      float* yp = Y + (size_t)(row0 + 8 * (lane >> 4)) * NOUT + c;
#pragma unroll
      for (int v = 0; v < 8; ++v) { *yp = yv[v]; yp += NOUT; }
    }
    if (doStats) {
      s += __shfl_down(s, 16);
      q += __shfl_down(q, 16);
      if (ok && lane < 16) {
        atomicAdd(statS + c, s);
        atomicAdd(statQ + c, q);
      }
    }
  }
}

// ---------------------------------------------------------------------------
// Edge kernel (PointGNNConv message path). D = feature dim (64 or 8).
// pass==1: layer1 of mlp_f, accumulate BN stats (LDS then global atomics).
// pass==2: layer1 + BN + ReLU -> LDS -> layer2 -> atomic-max scatter to agg.
// Features staged per 16-edge tile in LDS as [x_j(D) | rel(3) | pad].
// ---------------------------------------------------------------------------
template <int D>
__global__ __launch_bounds__(256) void k_edge(
    const f16* __restrict__ zf16, const float* __restrict__ pos,
    const float* __restrict__ delta, const int* __restrict__ src,
    const int* __restrict__ dst, const f16* __restrict__ Wp1,
    const float* __restrict__ b1, const float* __restrict__ scale1,
    const float* __restrict__ shift1, const f16* __restrict__ Wp2,
    const float* __restrict__ b2, float* __restrict__ statS,
    float* __restrict__ statQ, float* __restrict__ agg, int ntiles, int pass) {
  constexpr int WV  = 8;
  constexpr int KP1 = (D == 64) ? 96 : 32;   // padded input features (D+3)
  constexpr int KC1 = KP1 / 32;
  constexpr int K2P = (D == 64) ? 64 : 32;   // padded hidden dim
  constexpr int KC2 = K2P / 32;
  constexpr int NT  = (D + 15) / 16;
  constexpr int LDZ = (D == 64) ? 64 : 32;   // row stride of zf16

  __shared__ f16 sF[WV][16 * KP1];
  __shared__ f16 sH[WV][16 * K2P];
  __shared__ int sD[WV][16];
  __shared__ float sS[64], sQ[64];

  int wv = threadIdx.x >> 5, lane = threadIdx.x & 31;
  int i = lane & 15, half = lane >> 4;

  if (threadIdx.x < 64) { sS[threadIdx.x] = 0.f; sQ[threadIdx.x] = 0.f; }
  {
    v8h z;
#pragma unroll
    for (int t = 0; t < 8; ++t) z[t] = (f16)0.f;
    f16* p = &sF[0][0];
    for (int t = threadIdx.x * 8; t < WV * 16 * KP1; t += blockDim.x * 8)
      *(v8h*)(p + t) = z;
    p = &sH[0][0];
    for (int t = threadIdx.x * 8; t < WV * 16 * K2P; t += blockDim.x * 8)
      *(v8h*)(p + t) = z;
  }
  __syncthreads();

  int tile    = blockIdx.x * WV + wv;
  bool active = tile < ntiles;
  f16* F  = sF[wv];
  f16* Hh = sH[wv];

  if (active) {                       // stage features for 16 edges
    long long e = (long long)tile * 16 + i;
    int s = src[e], d = dst[e];
    if (half == 0) {
      sD[wv][i] = d;
#pragma unroll
      for (int j = 0; j < 3; ++j)
        F[i * KP1 + D + j] =
            (f16)(pos[(size_t)s * 3 + j] - pos[(size_t)d * 3 + j] +
                  delta[(size_t)d * 3 + j]);
    }
    const f16* xs = zf16 + (size_t)s * LDZ;
    if (D == 64) {
#pragma unroll
      for (int j = 0; j < 32; j += 8)
        *(v8h*)(F + i * KP1 + half * 32 + j) = *(const v8h*)(xs + half * 32 + j);
    } else {
      if (half == 0) *(v8h*)(F + i * KP1) = *(const v8h*)xs;
    }
  }
  __syncthreads();

  if (active) {                       // layer 1
    v16h A[KC1];
#pragma unroll
    for (int k = 0; k < KC1; ++k) A[k] = load_a(F, KP1, k * 32, lane);
#pragma unroll
    for (int nt = 0; nt < NT; ++nt) {
      v8f acc;
#pragma unroll
      for (int v = 0; v < 8; ++v) acc[v] = 0.f;
#pragma unroll
      for (int k = 0; k < KC1; ++k) {
        v16h B = *(const v16h*)(Wp1 + ((size_t)(nt * KC1 + k) * 32 + lane) * 16);
        acc = wmma_f16(A[k], B, acc);
      }
      int c   = nt * 16 + i;
      bool ok = (D % 16 == 0) ? true : (c < D);
      float bb = ok ? b1[c] : 0.f;
      if (pass == 1) {
        float s1 = 0.f, q1 = 0.f;
#pragma unroll
        for (int v = 0; v < 8; ++v) {
          float y = acc[v] + bb;
          s1 += y;
          q1 += y * y;
        }
        if (ok) { atomicAdd(&sS[c], s1); atomicAdd(&sQ[c], q1); }
      } else {
        float sc = ok ? scale1[c] : 0.f;
        float sh = ok ? shift1[c] : 0.f;
        if (ok) {
          f16* hp = Hh + (8 * half) * K2P + c;
#pragma unroll
          for (int v = 0; v < 8; ++v) {
            float h = fmaxf(0.f, (acc[v] + bb) * sc + sh);
            *hp = (f16)h;
            hp += K2P;
          }
        }
      }
    }
  }
  __syncthreads();

  if (pass == 2) {
    if (active) {                     // layer 2 + scatter-max
      v16h A2[KC2];
#pragma unroll
      for (int k = 0; k < KC2; ++k) A2[k] = load_a(Hh, K2P, k * 32, lane);
      int dr[8];
#pragma unroll
      for (int v = 0; v < 8; ++v) dr[v] = sD[wv][v + 8 * half];
#pragma unroll
      for (int nt = 0; nt < NT; ++nt) {
        v8f acc;
#pragma unroll
        for (int v = 0; v < 8; ++v) acc[v] = 0.f;
#pragma unroll
        for (int k = 0; k < KC2; ++k) {
          v16h B =
              *(const v16h*)(Wp2 + ((size_t)(nt * KC2 + k) * 32 + lane) * 16);
          acc = wmma_f16(A2[k], B, acc);
        }
        int c = nt * 16 + i;
        if ((D % 16 == 0) || c < D) {
          float bb = b2[c];
#pragma unroll
          for (int v = 0; v < 8; ++v) {
            float y = acc[v] + bb;
            atomic_fmax_f32(agg + (size_t)dr[v] * D + c, y);
          }
        }
      }
    }
  } else {                            // flush block-level stats
    if (threadIdx.x < 64) {
      atomicAdd(statS + threadIdx.x, sS[threadIdx.x]);
      atomicAdd(statQ + threadIdx.x, sQ[threadIdx.x]);
    }
  }
}

// ----------------------------- elementwise ---------------------------------
__global__ void k_fill(float* p, int n, float v) {
  int idx = blockIdx.x * blockDim.x + threadIdx.x;
  if (idx < n) p[idx] = v;
}

__global__ void k_fin(const float* S, const float* Q, float invn,
                      const float* gamma, const float* beta, float* scale,
                      float* shift, int n, float eps) {
  int c = threadIdx.x;
  if (c >= n) return;
  float mu  = S[c] * invn;
  float var = Q[c] * invn - mu * mu;
  float sc  = gamma[c] * rsqrtf(var + eps);
  scale[c] = sc;
  shift[c] = beta[c] - mu * sc;
}

// BN+ReLU apply; one v8h (16B) f16 store per thread. ld multiple of 8.
__global__ void k_bnrelu(const float* __restrict__ Y, int n, int cdim,
                         const float* __restrict__ scale,
                         const float* __restrict__ shift,
                         float* __restrict__ outF32, f16* __restrict__ outF16,
                         int ld) {
  int w   = ld >> 3;
  int idx = blockIdx.x * blockDim.x + threadIdx.x;
  if (idx >= n * w) return;
  int r = idx / w, c0 = (idx % w) * 8;
  v8h o;
#pragma unroll
  for (int j = 0; j < 8; ++j) {
    int c = c0 + j;
    float v = 0.f;
    if (c < cdim) {
      v = fmaxf(0.f, Y[(size_t)r * cdim + c] * scale[c] + shift[c]);
      if (outF32) outF32[(size_t)r * cdim + c] = v;
    }
    o[j] = (f16)v;
  }
  *(v8h*)(outF16 + (size_t)r * ld + c0) = o;
}

// f32 -> padded f16 convert; optional !isfinite -> 0 (for segment_max fix).
__global__ void k_cvtpad(const float* __restrict__ Y, int n, int cdim,
                         f16* __restrict__ out, int ld, int fix) {
  int w   = ld >> 3;
  int idx = blockIdx.x * blockDim.x + threadIdx.x;
  if (idx >= n * w) return;
  int r = idx / w, c0 = (idx % w) * 8;
  v8h o;
#pragma unroll
  for (int j = 0; j < 8; ++j) {
    int c = c0 + j;
    float v = (c < cdim) ? Y[(size_t)r * cdim + c] : 0.f;
    if (fix && !(fabsf(v) < 3.0e38f)) v = 0.f;
    o[j] = (f16)v;
  }
  *(v8h*)(out + (size_t)r * ld + c0) = o;
}

__global__ void k_resid(const float* __restrict__ zin,
                        const float* __restrict__ gy, float* __restrict__ zout,
                        int n, int cdim, float* __restrict__ S,
                        float* __restrict__ Q, int doStats) {
  int c = threadIdx.x;
  if (c >= cdim) return;
  int r0 = blockIdx.x * 256;
  int re = r0 + 256;
  if (re > n) re = n;
  float s = 0.f, q = 0.f;
  for (int r = r0; r < re; ++r) {
    size_t o = (size_t)r * cdim + c;
    float v  = zin[o] + gy[o];
    zout[o]  = v;
    s += v;
    q += v * v;
  }
  if (doStats) { atomicAdd(S + c, s); atomicAdd(Q + c, q); }
}

// ---------------------------------------------------------------------------
struct LinP { const float *W, *b, *g, *be; };
struct ConvP { LinP h[2], f[2], g[2]; };

extern "C" void kernel_launch(void* const* d_in, const int* in_sizes, int n_in,
                              void* d_out, int out_size, void* d_ws,
                              size_t ws_size, hipStream_t stream) {
  const float* x    = (const float*)d_in[0];
  const float* pos  = (const float*)d_in[1];
  const int*   ei   = (const int*)d_in[2];
  const int N = in_sizes[0] / 7;
  const int E = in_sizes[2] / 2;
  const int* srcI = ei;        // edge_index[0,:]
  const int* dstI = ei + E;    // edge_index[1,:]

  // ---- walk param leaves in setup_inputs() insertion order ----------------
  int li = 3;
  auto nf = [&]() { return (const float*)d_in[li++]; };
  auto rdLin = [&](bool bn) {
    LinP l;
    l.W = nf(); l.b = nf();
    l.g = bn ? nf() : nullptr;
    l.be = bn ? nf() : nullptr;
    return l;
  };
  auto rdConv = [&]() {
    ConvP c;
    c.h[0] = rdLin(true); c.h[1] = rdLin(false);
    c.f[0] = rdLin(true); c.f[1] = rdLin(false);
    c.g[0] = rdLin(true); c.g[1] = rdLin(false);
    return c;
  };
  LinP enc[3];
  enc[0] = rdLin(true); enc[1] = rdLin(true); enc[2] = rdLin(false);
  LinP itf = rdLin(false);
  ConvP convs[3];
  for (int i = 0; i < 3; ++i) convs[i] = rdConv();
  const float *bnG[3], *bnB[3];
  for (int i = 0; i < 3; ++i) { bnG[i] = nf(); bnB[i] = nf(); }
  LinP op = rdLin(false);
  ConvP oconv = rdConv();
  LinP dec[3];
  dec[0] = rdLin(true); dec[1] = rdLin(true); dec[2] = rdLin(false);

  // ---- workspace layout ---------------------------------------------------
  char* wp = (char*)d_ws;
  auto alloc = [&](size_t bytes) {
    void* p = wp;
    wp += (bytes + 255) & ~(size_t)255;
    return p;
  };
  float* zf32  = (float*)alloc((size_t)N * 64 * 4);
  f16*   zf16  = (f16*)alloc((size_t)N * 64 * 2);
  float* tmpY  = (float*)alloc((size_t)N * 64 * 4);
  f16*   tA    = (f16*)alloc((size_t)N * 64 * 2);
  f16*   tB    = (f16*)alloc((size_t)N * 64 * 2);
  float* delta = (float*)alloc((size_t)N * 3 * 4);
  float* agg   = (float*)alloc((size_t)N * 64 * 4);
  float* SQ    = (float*)alloc(128 * 4);
  float* S = SQ, *Q = SQ + 64;
  float* sc = (float*)alloc(64 * 4);
  float* sh = (float*)alloc(64 * 4);
  f16* packArena = (f16*)alloc(256 * 1024);
  f16* pcur = packArena;

  auto pack = [&](const float* W, int K, int Nout, int kc, int nt,
                  int perm) -> const f16* {
    f16* dp = pcur;
    int total = kc * nt * 512;
    pcur += total;
    k_pack<<<(total + 255) / 256, 256, 0, stream>>>(W, K, Nout, kc, nt, perm, dp);
    return dp;
  };

  const int waves = (N + 15) / 16;
  const int gblk  = (waves + 7) / 8;
  const int etl   = (E + 15) / 16;
  const int eblk  = (etl + 7) / 8;
  const float invN = 1.0f / (float)N;
  const float invE = 1.0f / (float)E;
  auto ew = [](int tot) { return (tot + 255) / 256; };
  auto zeroSQ = [&]() { k_fill<<<1, 128, 0, stream>>>(S, 128, 0.f); };

  // ---- pack all weights ---------------------------------------------------
  const f16* pe0 = pack(enc[0].W, 7, 64, 1, 4, 0);
  const f16* pe1 = pack(enc[1].W, 64, 64, 2, 4, 0);
  const f16* pe2 = pack(enc[2].W, 64, 8, 2, 1, 0);
  const f16* pit = pack(itf.W, 8, 64, 1, 4, 0);
  const f16 *pc[3][6];
  for (int i = 0; i < 3; ++i) {
    pc[i][0] = pack(convs[i].h[0].W, 64, 64, 2, 4, 0);
    pc[i][1] = pack(convs[i].h[1].W, 64, 3, 2, 1, 0);
    pc[i][2] = pack(convs[i].f[0].W, 67, 64, 3, 4, 64);  // perm: [x|rel]
    pc[i][3] = pack(convs[i].f[1].W, 64, 64, 2, 4, 0);
    pc[i][4] = pack(convs[i].g[0].W, 64, 64, 2, 4, 0);
    pc[i][5] = pack(convs[i].g[1].W, 64, 64, 2, 4, 0);
  }
  const f16* pop = pack(op.W, 64, 8, 2, 1, 0);
  const f16* po[6] = {
      pack(oconv.h[0].W, 8, 8, 1, 1, 0),  pack(oconv.h[1].W, 8, 3, 1, 1, 0),
      pack(oconv.f[0].W, 11, 8, 1, 1, 8), pack(oconv.f[1].W, 8, 8, 1, 1, 0),
      pack(oconv.g[0].W, 8, 8, 1, 1, 0),  pack(oconv.g[1].W, 8, 8, 1, 1, 0)};
  const f16* pd0 = pack(dec[0].W, 8, 64, 1, 4, 0);
  const f16* pd1 = pack(dec[1].W, 64, 64, 2, 4, 0);
  const f16* pd2 = pack(dec[2].W, 64, 4, 2, 1, 0);

  // ---- encoder + input transform -----------------------------------------
  k_cvtpad<<<ew(N * 4), 256, 0, stream>>>(x, N, 7, tA, 32, 0);
  zeroSQ();
  k_gemm<1, 4, 64><<<gblk, 256, 0, stream>>>(tA, pe0, enc[0].b, tmpY, N, S, Q, 1);
  k_fin<<<1, 64, 0, stream>>>(S, Q, invN, enc[0].g, enc[0].be, sc, sh, 64, 1e-5f);
  k_bnrelu<<<ew(N * 8), 256, 0, stream>>>(tmpY, N, 64, sc, sh, nullptr, tB, 64);
  zeroSQ();
  k_gemm<2, 4, 64><<<gblk, 256, 0, stream>>>(tB, pe1, enc[1].b, tmpY, N, S, Q, 1);
  k_fin<<<1, 64, 0, stream>>>(S, Q, invN, enc[1].g, enc[1].be, sc, sh, 64, 1e-5f);
  k_bnrelu<<<ew(N * 8), 256, 0, stream>>>(tmpY, N, 64, sc, sh, nullptr, tA, 64);
  k_gemm<2, 1, 8><<<gblk, 256, 0, stream>>>(tA, pe2, enc[2].b, tmpY, N, S, Q, 0);
  k_cvtpad<<<ew(N * 4), 256, 0, stream>>>(tmpY, N, 8, tB, 32, 0);
  k_gemm<1, 4, 64><<<gblk, 256, 0, stream>>>(tB, pit, itf.b, zf32, N, S, Q, 0);
  k_cvtpad<<<ew(N * 8), 256, 0, stream>>>(zf32, N, 64, zf16, 64, 0);

  // ---- PointGNN conv ------------------------------------------------------
  auto run_conv = [&](int d, const ConvP& C, const f16* ph1, const f16* ph2,
                      const f16* pf1, const f16* pf2, const f16* pg1,
                      const f16* pg2, const float* obG, const float* obB) {
    int ldz = (d == 64) ? 64 : 32;
    // h-MLP -> delta [N,3]
    zeroSQ();
    if (d == 64)
      k_gemm<2, 4, 64><<<gblk, 256, 0, stream>>>(zf16, ph1, C.h[0].b, tmpY, N, S, Q, 1);
    else
      k_gemm<1, 1, 8><<<gblk, 256, 0, stream>>>(zf16, ph1, C.h[0].b, tmpY, N, S, Q, 1);
    k_fin<<<1, 64, 0, stream>>>(S, Q, invN, C.h[0].g, C.h[0].be, sc, sh, d, 1e-5f);
    k_bnrelu<<<ew(N * (ldz / 8)), 256, 0, stream>>>(tmpY, N, d, sc, sh, nullptr, tA, ldz);
    if (d == 64)
      k_gemm<2, 1, 3><<<gblk, 256, 0, stream>>>(tA, ph2, C.h[1].b, delta, N, S, Q, 0);
    else
      k_gemm<1, 1, 3><<<gblk, 256, 0, stream>>>(tA, ph2, C.h[1].b, delta, N, S, Q, 0);
    // edge pass 1: stats of f-hidden over all E edges
    zeroSQ();
    if (d == 64)
      k_edge<64><<<eblk, 256, 0, stream>>>(zf16, pos, delta, srcI, dstI, pf1,
                                           C.f[0].b, nullptr, nullptr, nullptr,
                                           nullptr, S, Q, nullptr, etl, 1);
    else
      k_edge<8><<<eblk, 256, 0, stream>>>(zf16, pos, delta, srcI, dstI, pf1,
                                          C.f[0].b, nullptr, nullptr, nullptr,
                                          nullptr, S, Q, nullptr, etl, 1);
    k_fin<<<1, 64, 0, stream>>>(S, Q, invE, C.f[0].g, C.f[0].be, sc, sh, d, 1e-5f);
    // edge pass 2: BN+ReLU, layer2, scatter-max into agg
    k_fill<<<ew(N * d), 256, 0, stream>>>(agg, N * d, -__builtin_huge_valf());
    if (d == 64)
      k_edge<64><<<eblk, 256, 0, stream>>>(zf16, pos, delta, srcI, dstI, pf1,
                                           C.f[0].b, sc, sh, pf2, C.f[1].b,
                                           nullptr, nullptr, agg, etl, 2);
    else
      k_edge<8><<<eblk, 256, 0, stream>>>(zf16, pos, delta, srcI, dstI, pf1,
                                          C.f[0].b, sc, sh, pf2, C.f[1].b,
                                          nullptr, nullptr, agg, etl, 2);
    k_cvtpad<<<ew(N * (ldz / 8)), 256, 0, stream>>>(agg, N, d, tA, ldz, 1);
    // g-MLP + residual (+ optional outer BN/ReLU)
    zeroSQ();
    if (d == 64)
      k_gemm<2, 4, 64><<<gblk, 256, 0, stream>>>(tA, pg1, C.g[0].b, tmpY, N, S, Q, 1);
    else
      k_gemm<1, 1, 8><<<gblk, 256, 0, stream>>>(tA, pg1, C.g[0].b, tmpY, N, S, Q, 1);
    k_fin<<<1, 64, 0, stream>>>(S, Q, invN, C.g[0].g, C.g[0].be, sc, sh, d, 1e-5f);
    k_bnrelu<<<ew(N * (ldz / 8)), 256, 0, stream>>>(tmpY, N, d, sc, sh, nullptr, tB, ldz);
    if (d == 64)
      k_gemm<2, 4, 64><<<gblk, 256, 0, stream>>>(tB, pg2, C.g[1].b, tmpY, N, S, Q, 0);
    else
      k_gemm<1, 1, 8><<<gblk, 256, 0, stream>>>(tB, pg2, C.g[1].b, tmpY, N, S, Q, 0);
    zeroSQ();
    k_resid<<<(N + 255) / 256, 64, 0, stream>>>(zf32, tmpY, zf32, N, d, S, Q,
                                                obG ? 1 : 0);
    if (obG) {
      k_fin<<<1, 64, 0, stream>>>(S, Q, invN, obG, obB, sc, sh, d, 1e-5f);
      k_bnrelu<<<ew(N * (ldz / 8)), 256, 0, stream>>>(zf32, N, d, sc, sh, zf32,
                                                      zf16, ldz);
    }
  };

  for (int i = 0; i < 3; ++i)
    run_conv(64, convs[i], pc[i][0], pc[i][1], pc[i][2], pc[i][3], pc[i][4],
             pc[i][5], bnG[i], bnB[i]);

  // ---- out projection + out conv (d=8) ------------------------------------
  k_gemm<2, 1, 8><<<gblk, 256, 0, stream>>>(zf16, pop, op.b, zf32, N, S, Q, 0);
  k_cvtpad<<<ew(N * 4), 256, 0, stream>>>(zf32, N, 8, zf16, 32, 0);
  run_conv(8, oconv, po[0], po[1], po[2], po[3], po[4], po[5], nullptr, nullptr);
  k_cvtpad<<<ew(N * 4), 256, 0, stream>>>(zf32, N, 8, zf16, 32, 0);

  // ---- decoder -------------------------------------------------------------
  zeroSQ();
  k_gemm<1, 4, 64><<<gblk, 256, 0, stream>>>(zf16, pd0, dec[0].b, tmpY, N, S, Q, 1);
  k_fin<<<1, 64, 0, stream>>>(S, Q, invN, dec[0].g, dec[0].be, sc, sh, 64, 1e-5f);
  k_bnrelu<<<ew(N * 8), 256, 0, stream>>>(tmpY, N, 64, sc, sh, nullptr, tA, 64);
  zeroSQ();
  k_gemm<2, 4, 64><<<gblk, 256, 0, stream>>>(tA, pd1, dec[1].b, tmpY, N, S, Q, 1);
  k_fin<<<1, 64, 0, stream>>>(S, Q, invN, dec[1].g, dec[1].be, sc, sh, 64, 1e-5f);
  k_bnrelu<<<ew(N * 8), 256, 0, stream>>>(tmpY, N, 64, sc, sh, nullptr, tB, 64);
  k_gemm<2, 1, 4><<<gblk, 256, 0, stream>>>(tB, pd2, dec[2].b, (float*)d_out, N,
                                            S, Q, 0);
}
